// SelfAttention_463856468461
// MI455X (gfx1250) — matrix operand order repaired and checked by reading, not compiled, not run on hardware
//
#include <hip/hip_runtime.h>
#include <hip/hip_bf16.h>

typedef __attribute__((ext_vector_type(16))) _Float16 v16h;
typedef __attribute__((ext_vector_type(8)))  _Float16 v8h;
typedef __attribute__((ext_vector_type(8)))  float    v8f;
typedef __attribute__((ext_vector_type(4)))  unsigned u32x4;
typedef __attribute__((ext_vector_type(8)))  unsigned u32x8;

#define SLEN 2048
#define DEMB 1024
#define NH   16
#define DH   64
#define MROWS 4096   // B*S

union V16 { v16h v; v8h h[2]; };

__device__ __forceinline__ v8f wmma32(v16h a, v16h b, v8f c) {
  return __builtin_amdgcn_wmma_f32_16x16x32_f16(false, a, false, b, (short)0, c,
                                                false, false);
}

// uniform (scalar-branch) test for "this is wave 0 of the block"
__device__ __forceinline__ bool wave0_uniform() {
  return (__builtin_amdgcn_readfirstlane((int)threadIdx.x) >> 5) == 0;
}

// A-fragment (16x32 f16), row-major source, leading dim ld (halfs).
__device__ __forceinline__ v16h load_a16g(const _Float16* base, int ld, int k0) {
  int lane = threadIdx.x & 31;
  int m = lane & 15, hi = lane >> 4;
  const _Float16* r = base + (size_t)m * ld + k0 + hi * 16;
  V16 u;
  u.h[0] = *(const v8h*)(r);
  u.h[1] = *(const v8h*)(r + 8);
  return u.v;
}

// w0 words for D# group1: data_size=2B, pad_enable, pad every 2^(iv+1) dwords by (am+1) dwords
#define TDM_W0_ROW16 0x06D10000u  // 16-dword rows (32 halfs) + 4-dword pad -> stride 40 halfs
#define TDM_W0_ROW32 0x07110000u  // 32-dword rows (64 halfs) + 4-dword pad -> stride 72 halfs

// Issue a 2-D TDM tile load: tile_d1 rows of tile_d0 halfs, row stride 'stride0' halfs.
__device__ __forceinline__ void tdm_load_2d(unsigned lds_off, const void* gaddr,
                                            unsigned tile_d0, unsigned tile_d1,
                                            unsigned stride0, unsigned w0pad) {
  unsigned long long ga = (unsigned long long)gaddr;
  u32x4 g0;
  g0[0] = 1u;                                   // count=1, user mode
  g0[1] = lds_off;                              // lds_addr
  g0[2] = (unsigned)ga;                         // global_addr[31:0]
  g0[3] = (unsigned)((ga >> 32) & 0x01FFFFFFu) | (2u << 30);  // ga[56:32] | type=2
  u32x8 g1;
  g1[0] = w0pad;                                // wg_mask=0 | data_size=1(2B) | pad cfg
  g1[1] = (tile_d0 & 0xFFFFu) << 16;            // tensor_dim0 = tile_d0 (no OOB)
  g1[2] = (tile_d0 >> 16) | ((tile_d1 & 0xFFFFu) << 16);
  g1[3] = (tile_d1 >> 16) | (tile_d0 << 16);    // td1 hi | tile_dim0
  g1[4] = tile_d1;                              // tile_dim1 | tile_dim2=0
  g1[5] = stride0;                              // tensor_dim0_stride[31:0]
  g1[6] = 0u;
  g1[7] = 0u;
  asm volatile("tensor_load_to_lds %0, %1" :: "s"(g0), "s"(g1) : "memory");
}

// ---------------------------------------------------------------- convert f32 -> f16
__global__ __launch_bounds__(256) void cvt_kernel(const float* __restrict__ s,
                                                  _Float16* __restrict__ d, int n) {
  for (int i = blockIdx.x * blockDim.x + threadIdx.x; i < n;
       i += gridDim.x * blockDim.x)
    d[i] = (_Float16)s[i];
}

// ----------------------------------------------------------- shared GEMM K-loop core
// Block: 256 thr (8 waves). Each wave: 2 row tiles (32 rows) x 64 cols.
// W tile (64 rows x 32 halfs) TDM-staged, double-buffered, LDS row stride 40 halfs.
__device__ __forceinline__ void gemm_core(const _Float16* a0base,
                                          const _Float16* a1base,
                                          const _Float16* W, int n0,
                                          _Float16* wt, unsigned ldsbase,
                                          v8f acc[2][4]) {
  const int WT_BUF = 64 * 40;                    // halfs per buffer
  const int NK = DEMB / 32;                      // 32 k-steps
  bool w0 = wave0_uniform();
  int lane = threadIdx.x & 31;
  int nl = lane & 15, hi = lane >> 4;

  if (w0) {
    tdm_load_2d(ldsbase,              W + (size_t)n0 * DEMB + 0,  32, 64, DEMB, TDM_W0_ROW16);
    tdm_load_2d(ldsbase + WT_BUF * 2, W + (size_t)n0 * DEMB + 32, 32, 64, DEMB, TDM_W0_ROW16);
  }
  v16h a0 = load_a16g(a0base, DEMB, 0);
  v16h a1 = load_a16g(a1base, DEMB, 0);

  for (int i = 0; i < NK - 2; i++) {             // steady state: branch-free body
    if (w0) __builtin_amdgcn_s_wait_tensorcnt(1);
    __syncthreads();                             // buffer i ready
    v16h a0n = load_a16g(a0base, DEMB, (i + 1) * 32);
    v16h a1n = load_a16g(a1base, DEMB, (i + 1) * 32);
    const _Float16* wb = wt + (i & 1) * WT_BUF;
#pragma unroll
    for (int j = 0; j < 4; j++) {
      v16h b = *(const v16h*)(wb + (j * 16 + nl) * 40 + hi * 16);
      acc[0][j] = wmma32(a0, b, acc[0][j]);
      acc[1][j] = wmma32(a1, b, acc[1][j]);
    }
    a0 = a0n; a1 = a1n;
    __syncthreads();                             // buffer i consumed
    if (w0)
      tdm_load_2d(ldsbase + (i & 1) * WT_BUF * 2,
                  W + (size_t)n0 * DEMB + (i + 2) * 32, 32, 64, DEMB, TDM_W0_ROW16);
  }
  {                                              // i = NK-2
    if (w0) __builtin_amdgcn_s_wait_tensorcnt(1);
    __syncthreads();
    v16h a0n = load_a16g(a0base, DEMB, (NK - 1) * 32);
    v16h a1n = load_a16g(a1base, DEMB, (NK - 1) * 32);
    const _Float16* wb = wt + ((NK - 2) & 1) * WT_BUF;
#pragma unroll
    for (int j = 0; j < 4; j++) {
      v16h b = *(const v16h*)(wb + (j * 16 + nl) * 40 + hi * 16);
      acc[0][j] = wmma32(a0, b, acc[0][j]);
      acc[1][j] = wmma32(a1, b, acc[1][j]);
    }
    a0 = a0n; a1 = a1n;
  }
  {                                              // i = NK-1
    if (w0) __builtin_amdgcn_s_wait_tensorcnt(0);
    __syncthreads();
    const _Float16* wb = wt + ((NK - 1) & 1) * WT_BUF;
#pragma unroll
    for (int j = 0; j < 4; j++) {
      v16h b = *(const v16h*)(wb + (j * 16 + nl) * 40 + hi * 16);
      acc[0][j] = wmma32(a0, b, acc[0][j]);
      acc[1][j] = wmma32(a1, b, acc[1][j]);
    }
  }
}

// --------------------------------------------------- Q/K/V projection (+RoPE on Q,K)
// grid: (D/64, M/256, 3)  block: 256
__global__ __launch_bounds__(256) void qkv_rope_kernel(
    const _Float16* __restrict__ xh,
    const _Float16* __restrict__ wqh, const _Float16* __restrict__ wkh,
    const _Float16* __restrict__ wvh,
    const float* __restrict__ bq, const float* __restrict__ bk,
    const float* __restrict__ bv,
    _Float16* __restrict__ qo, _Float16* __restrict__ ko,
    _Float16* __restrict__ vto) {
  extern __shared__ char smem[];                 // 10240 B
  _Float16* wt = (_Float16*)smem;
  unsigned ldsbase = (unsigned)(unsigned long long)(void*)smem;

  int z = blockIdx.z;
  const _Float16* W    = (z == 0) ? wqh : (z == 1) ? wkh : wvh;
  const float*    bias = (z == 0) ? bq  : (z == 1) ? bk  : bv;

  int wave = threadIdx.x >> 5, lane = threadIdx.x & 31;
  int m0 = blockIdx.y * 256 + wave * 32;
  int n0 = blockIdx.x * 64;
  int nl = lane & 15, hi = lane >> 4;

  v8f zero = {0.f, 0.f, 0.f, 0.f, 0.f, 0.f, 0.f, 0.f};
  v8f acc[2][4];
#pragma unroll
  for (int t = 0; t < 2; t++)
#pragma unroll
    for (int j = 0; j < 4; j++) acc[t][j] = zero;

  gemm_core(xh + (size_t)m0 * DEMB, xh + (size_t)(m0 + 16) * DEMB, W, n0, wt,
            ldsbase, acc);

  _Float16* qkdst = (z == 0) ? qo : ko;
#pragma unroll
  for (int t = 0; t < 2; t++) {
#pragma unroll
    for (int j = 0; j < 4; j++) {
      int e = n0 + j * 16 + nl;
      float bval = bias[e];
      int hh = e >> 6, dh = e & 63;
#pragma unroll
      for (int r = 0; r < 8; r++) {
        int m = m0 + t * 16 + r + 8 * hi;
        int bidx = m >> 11;                      // m / SLEN
        int s = m & (SLEN - 1);
        float val = acc[t][j][r] + bval;
        if (z < 2) {                             // RoPE
          float other = __shfl_xor(val, 1, 32);
          int p2 = dh & ~1;
          float ang = (float)s * __expf(-(float)p2 * 0.14391156831f);
          float c = __cosf(ang), sn = __sinf(ang);
          val = (dh & 1) ? (val * c + other * sn) : (val * c - other * sn);
          qkdst[(((size_t)bidx * NH + hh) * SLEN + s) * DH + dh] = (_Float16)val;
        } else {                                 // V transposed: [B,H,Dh,S]
          vto[(((size_t)bidx * NH + hh) * DH + dh) * SLEN + s] = (_Float16)val;
        }
      }
    }
  }
}

// --------------------------------------------------------- flash attention per (b,h)
// grid: (S/64, B*H)  block: 128 (4 waves x 16 query rows)
__global__ __launch_bounds__(128) void attn_kernel(
    const _Float16* __restrict__ q, const _Float16* __restrict__ k,
    const _Float16* __restrict__ vt, _Float16* __restrict__ attnH) {
  extern __shared__ char smem[];
  _Float16* pls = (_Float16*)smem;               // [4][16][40]  (5120 B)
  const unsigned KOFF = 5120;                    // ktile [2][32][72] (9216 B)
  const unsigned VOFF = 14336;                   // vtile [2][64][40] (10240 B)
  _Float16* kt0 = (_Float16*)(smem + KOFF);
  _Float16* vt0 = (_Float16*)(smem + VOFF);
  const int KT_BUF = 32 * 72, VT_BUF = 64 * 40;  // halfs
  unsigned ldsbase = (unsigned)(unsigned long long)(void*)smem;

  int bh = blockIdx.y;
  int b = bh >> 4, h = bh & 15;
  const _Float16* qb = q  + (size_t)bh * SLEN * DH;
  const _Float16* kb = k  + (size_t)bh * SLEN * DH;
  const _Float16* vb = vt + (size_t)bh * DH * SLEN;

  int wave = threadIdx.x >> 5, lane = threadIdx.x & 31;
  int nl = lane & 15, hi = lane >> 4;
  int qrow0 = blockIdx.x * 64 + wave * 16;
  bool w0 = wave0_uniform();

  v16h qa0 = load_a16g(qb + (size_t)qrow0 * DH, DH, 0);
  v16h qa1 = load_a16g(qb + (size_t)qrow0 * DH, DH, 32);

  const int NT = SLEN / 32;                      // 64 key tiles
  if (w0) {
#pragma unroll
    for (int i = 0; i < 2; i++) {
      tdm_load_2d(ldsbase + KOFF + (i & 1) * KT_BUF * 2, kb + (size_t)i * 32 * DH,
                  64, 32, DH, TDM_W0_ROW32);
      tdm_load_2d(ldsbase + VOFF + (i & 1) * VT_BUF * 2, vb + i * 32,
                  32, 64, SLEN, TDM_W0_ROW16);
    }
  }

  v8f zero = {0.f, 0.f, 0.f, 0.f, 0.f, 0.f, 0.f, 0.f};
  v8f o[4];
#pragma unroll
  for (int j = 0; j < 4; j++) o[j] = zero;
  float mx[8], l[8], al[8];
#pragma unroll
  for (int r = 0; r < 8; r++) { mx[r] = -3.0e38f; l[r] = 0.f; }

  _Float16* pw = pls + wave * 640;

  auto tile_body = [&](int i) {
    const _Float16* kbuf = kt0 + (i & 1) * KT_BUF;
    const _Float16* vbuf = vt0 + (i & 1) * VT_BUF;

    v8f sc0 = zero, sc1 = zero;
    sc0 = wmma32(qa0, *(const v16h*)(kbuf + nl * 72 + 0  + hi * 16), sc0);
    sc0 = wmma32(qa1, *(const v16h*)(kbuf + nl * 72 + 32 + hi * 16), sc0);
    sc1 = wmma32(qa0, *(const v16h*)(kbuf + (16 + nl) * 72 + 0  + hi * 16), sc1);
    sc1 = wmma32(qa1, *(const v16h*)(kbuf + (16 + nl) * 72 + 32 + hi * 16), sc1);

#pragma unroll
    for (int r = 0; r < 8; r++) {
      float v0 = sc0[r] * 0.125f, v1 = sc1[r] * 0.125f;   // 1/sqrt(64)
      float tm = fmaxf(v0, v1);
      tm = fmaxf(tm, __shfl_xor(tm, 1, 32));
      tm = fmaxf(tm, __shfl_xor(tm, 2, 32));
      tm = fmaxf(tm, __shfl_xor(tm, 4, 32));
      tm = fmaxf(tm, __shfl_xor(tm, 8, 32));
      float nm = fmaxf(mx[r], tm);
      float alpha = __expf(mx[r] - nm);
      float p0 = __expf(v0 - nm), p1 = __expf(v1 - nm);
      float rs = p0 + p1;
      rs += __shfl_xor(rs, 1, 32);
      rs += __shfl_xor(rs, 2, 32);
      rs += __shfl_xor(rs, 4, 32);
      rs += __shfl_xor(rs, 8, 32);
      l[r] = l[r] * alpha + rs;
      mx[r] = nm; al[r] = alpha;
      pw[(r + 8 * hi) * 40 + nl]      = (_Float16)p0;
      pw[(r + 8 * hi) * 40 + nl + 16] = (_Float16)p1;
    }
#pragma unroll
    for (int j = 0; j < 4; j++)
#pragma unroll
      for (int r = 0; r < 8; r++) o[j][r] *= al[r];

    __syncthreads();                             // P written
    V16 u;
    u.h[0] = *(const v8h*)(pw + nl * 40 + hi * 16);
    u.h[1] = *(const v8h*)(pw + nl * 40 + hi * 16 + 8);
    v16h pa = u.v;
#pragma unroll
    for (int j = 0; j < 4; j++) {
      v16h vfrag = *(const v16h*)(vbuf + (j * 16 + nl) * 40 + hi * 16);
      o[j] = wmma32(pa, vfrag, o[j]);
    }
  };

  for (int i = 0; i < NT - 2; i++) {             // steady state
    if (w0) __builtin_amdgcn_s_wait_tensorcnt(2);
    __syncthreads();                             // K/V tile i ready
    tile_body(i);
    __syncthreads();                             // tile i consumed
    if (w0) {
      tdm_load_2d(ldsbase + KOFF + (i & 1) * KT_BUF * 2,
                  kb + (size_t)(i + 2) * 32 * DH, 64, 32, DH, TDM_W0_ROW32);
      tdm_load_2d(ldsbase + VOFF + (i & 1) * VT_BUF * 2,
                  vb + (i + 2) * 32, 32, 64, SLEN, TDM_W0_ROW16);
    }
  }
  {                                              // i = NT-2
    if (w0) __builtin_amdgcn_s_wait_tensorcnt(2);
    __syncthreads();
    tile_body(NT - 2);
    __syncthreads();
  }
  {                                              // i = NT-1
    if (w0) __builtin_amdgcn_s_wait_tensorcnt(0);
    __syncthreads();
    tile_body(NT - 1);
  }

#pragma unroll
  for (int r = 0; r < 8; r++) {
    int s = qrow0 + r + 8 * hi;
    float inv = 1.0f / l[r];
#pragma unroll
    for (int j = 0; j < 4; j++) {
      int e = h * DH + j * 16 + nl;
      attnH[((size_t)b * SLEN + s) * DEMB + e] = (_Float16)(o[j][r] * inv);
    }
  }
}

// ------------------------------------------------------------- output projection
// grid: (D/64, M/256)  block: 256
__global__ __launch_bounds__(256) void oproj_kernel(
    const _Float16* __restrict__ ah, const _Float16* __restrict__ woh,
    const float* __restrict__ bo, float* __restrict__ out) {
  extern __shared__ char smem[];
  _Float16* wt = (_Float16*)smem;
  unsigned ldsbase = (unsigned)(unsigned long long)(void*)smem;

  int wave = threadIdx.x >> 5, lane = threadIdx.x & 31;
  int m0 = blockIdx.y * 256 + wave * 32;
  int n0 = blockIdx.x * 64;
  int nl = lane & 15, hi = lane >> 4;

  v8f zero = {0.f, 0.f, 0.f, 0.f, 0.f, 0.f, 0.f, 0.f};
  v8f acc[2][4];
#pragma unroll
  for (int t = 0; t < 2; t++)
#pragma unroll
    for (int j = 0; j < 4; j++) acc[t][j] = zero;

  gemm_core(ah + (size_t)m0 * DEMB, ah + (size_t)(m0 + 16) * DEMB, woh, n0, wt,
            ldsbase, acc);

#pragma unroll
  for (int t = 0; t < 2; t++) {
#pragma unroll
    for (int j = 0; j < 4; j++) {
      int e = n0 + j * 16 + nl;
      float bval = bo[e];
#pragma unroll
      for (int r = 0; r < 8; r++) {
        int m = m0 + t * 16 + r + 8 * hi;
        out[(size_t)m * DEMB + e] = acc[t][j][r] + bval;
      }
    }
  }
}

extern "C" void kernel_launch(void* const* d_in, const int* in_sizes, int n_in,
                              void* d_out, int out_size, void* d_ws, size_t ws_size,
                              hipStream_t stream) {
  const float* x  = (const float*)d_in[0];
  const float* Wq = (const float*)d_in[1];
  const float* bq = (const float*)d_in[2];
  const float* Wk = (const float*)d_in[3];
  const float* bk = (const float*)d_in[4];
  const float* Wv = (const float*)d_in[5];
  const float* bv = (const float*)d_in[6];
  const float* Wo = (const float*)d_in[7];
  const float* bo = (const float*)d_in[8];
  float* out = (float*)d_out;

  char* p = (char*)d_ws;
  auto carve = [&](size_t nhalfs) -> _Float16* {
    _Float16* r = (_Float16*)p;
    p += ((nhalfs * 2 + 255) & ~(size_t)255);
    return r;
  };
  _Float16* xh  = carve((size_t)MROWS * DEMB);
  _Float16* wqh = carve((size_t)DEMB * DEMB);
  _Float16* wkh = carve((size_t)DEMB * DEMB);
  _Float16* wvh = carve((size_t)DEMB * DEMB);
  _Float16* woh = carve((size_t)DEMB * DEMB);
  _Float16* qf  = carve((size_t)MROWS * DEMB);   // [B,H,S,Dh]
  _Float16* kf  = carve((size_t)MROWS * DEMB);   // [B,H,S,Dh]
  _Float16* vtf = carve((size_t)MROWS * DEMB);   // [B,H,Dh,S]
  _Float16* ath = carve((size_t)MROWS * DEMB);   // [M, D]

  cvt_kernel<<<1024, 256, 0, stream>>>(x,  xh,  MROWS * DEMB);
  cvt_kernel<<<512,  256, 0, stream>>>(Wq, wqh, DEMB * DEMB);
  cvt_kernel<<<512,  256, 0, stream>>>(Wk, wkh, DEMB * DEMB);
  cvt_kernel<<<512,  256, 0, stream>>>(Wv, wvh, DEMB * DEMB);
  cvt_kernel<<<512,  256, 0, stream>>>(Wo, woh, DEMB * DEMB);

  qkv_rope_kernel<<<dim3(DEMB / 64, MROWS / 256, 3), 256, 10240, stream>>>(
      xh, wqh, wkh, wvh, bq, bk, bv, qf, kf, vtf);

  attn_kernel<<<dim3(SLEN / 64, 2 * NH), 128, 24576, stream>>>(qf, kf, vtf, ath);

  oproj_kernel<<<dim3(DEMB / 64, MROWS / 256), 256, 10240, stream>>>(ath, woh, bo, out);
}
